// StructureAwareCrossAttention_69922067578975
// MI455X (gfx1250) — matrix operand order repaired and checked by reading, not compiled, not run on hardware
//
#include <hip/hip_runtime.h>
#include <hip/hip_bf16.h>
#include <math.h>

// ---------------------------------------------------------------------------
// StructureAwareCrossAttention for MI455X (gfx1250, wave32, WMMA + TDM)
//   B=4, N=2048, D_MODEL=512, NHEAD=8, D_HEAD=64
// fp32->bf16 convert, WMMA projections, flash-attention with online softmax
// (bf16 WMMA, f32 accumulate, K/V tiles staged by the Tensor Data Mover,
// multiplicative sigmoid gate in exp2 domain), WMMA output projection.
// ---------------------------------------------------------------------------

#define BB    4
#define NSEQ  2048
#define DM    512
#define HEADS 8
#define DH    64
#define SCALEF 0.125f          // 1/sqrt(64)
#define GEPS   1e-8f
#define LOG2E  1.4426950408889634f

typedef __attribute__((ext_vector_type(16))) __bf16 v16bf;
typedef __attribute__((ext_vector_type(8)))  __bf16 v8bf;
typedef __attribute__((ext_vector_type(8)))  float  v8f;
typedef __attribute__((ext_vector_type(4)))  unsigned int u32x4;
typedef __attribute__((ext_vector_type(4)))  int  i32x4;
typedef __attribute__((ext_vector_type(8)))  int  i32x8;

__device__ __forceinline__ float fast_rcp(float x) { return __builtin_amdgcn_rcpf(x); }
__device__ __forceinline__ float fast_exp2(float x) { return __builtin_amdgcn_exp2f(x); }

__device__ __forceinline__ v16bf join16(v8bf lo, v8bf hi) {
  v16bf r;
#pragma unroll
  for (int i = 0; i < 8; ++i) { r[i] = lo[i]; r[i + 8] = hi[i]; }
  return r;
}

// A-fragment (16x32, MxK), bf16. ISA layout: lanes 0-15 row M=lane,
// elements 0..7 -> K = half*8.. ; elements 8..15 -> K = 16+half*8.. .
__device__ __forceinline__ v16bf load_afrag(const __bf16* base, int ld, int lane) {
  const int row = lane & 15, half = lane >> 4;
  const __bf16* p = base + (size_t)row * ld + half * 8;
  v8bf lo = *(const v8bf*)p;
  v8bf hi = *(const v8bf*)(p + 16);
  return join16(lo, hi);
}

// B-fragment (32x16, KxN) from a source stored as Bsrc[n][k] (k contiguous):
// lane n = lane&15, elements 0..7 -> k = (lane>>4)*8 + j; 8..15 -> k = 16 + (lane>>4)*8 + (j-8).
__device__ __forceinline__ v16bf load_bfrag_t(const __bf16* base, int ld, int lane) {
  const int n = lane & 15, khalf = lane >> 4;
  const __bf16* p = base + (size_t)n * ld + khalf * 8;
  v8bf lo = *(const v8bf*)p;
  v8bf hi = *(const v8bf*)(p + 16);
  return join16(lo, hi);
}

__device__ __forceinline__ v8f wmma_bf16(v16bf a, v16bf b, v8f c) {
  return __builtin_amdgcn_wmma_f32_16x16x32_bf16(
      false, a, false, b, (short)0, c, false, false);
}

// ---------------------------------------------------------------------------
// TDM: 2D tile load global->LDS (bf16 elements). Descriptor per ISA §8.3/8.4.
// 6-arg builtin (amdgpu-toolchain / clang-23).
// ---------------------------------------------------------------------------
__device__ __forceinline__ void tdm_load_tile_2d(unsigned int lds_addr,
                                                 const void* gptr,
                                                 unsigned int tile_d0,
                                                 unsigned int tile_d1,
                                                 unsigned int stride_d0) {
  const unsigned long long ga = (unsigned long long)gptr;
  u32x4 g0;
  g0[0] = 1u;                                           // count=1, user mode
  g0[1] = lds_addr;                                     // LDS byte offset
  g0[2] = (unsigned int)ga;                             // global_addr[31:0]
  g0[3] = (unsigned int)((ga >> 32) & 0x01FFFFFFu) | (2u << 30);  // [56:32]|type=2
  i32x8 g1;
  g1[0] = (int)(1u << 16);                              // data_size = 2 bytes
  g1[1] = (int)(tile_d0 << 16);                         // tensor_dim0 lo @ bit48
  g1[2] = (int)((tile_d0 >> 16) | (tile_d1 << 16));     // dim0 hi | tensor_dim1 lo
  g1[3] = (int)((tile_d1 >> 16) | (tile_d0 << 16));     // dim1 hi | tile_dim0
  g1[4] = (int)(tile_d1 & 0xFFFFu);                     // tile_dim1 (tile_dim2=0)
  g1[5] = (int)stride_d0;                               // tensor_dim0_stride lo
  g1[6] = 0;
  g1[7] = 0;
  i32x4 z4;
  z4[0] = 0; z4[1] = 0; z4[2] = 0; z4[3] = 0;
  i32x8 z8;
#pragma unroll
  for (int i = 0; i < 8; ++i) z8[i] = 0;
  __builtin_amdgcn_tensor_load_to_lds(g0, g1, z4, z4, z8, 0);
}

// ---------------------------------------------------------------------------
// fp32 -> bf16 elementwise
// ---------------------------------------------------------------------------
__global__ void cvt_f32_bf16(const float* __restrict__ src,
                             __bf16* __restrict__ dst, int n) {
  int i = blockIdx.x * blockDim.x + threadIdx.x;
  int stride = gridDim.x * blockDim.x;
  for (; i < n; i += stride) dst[i] = (__bf16)src[i];
}

// ---------------------------------------------------------------------------
// Y = X * W^T + bias.  X:[M,K] bf16, W:[N,K] bf16, bias:[N] f32.
// Block = 256 threads (8 waves); each wave owns a 16x64 tile of Y.
// ---------------------------------------------------------------------------
template <bool OUT_F32>
__global__ void __launch_bounds__(256)
gemm_xwT_wmma(const __bf16* __restrict__ X, const __bf16* __restrict__ W,
              const float* __restrict__ bias, void* __restrict__ Yv,
              int M, int Nn, int K) {
  const int lane = threadIdx.x & 31;
  const int wave = threadIdx.x >> 5;
  const int m0 = blockIdx.x * 128 + wave * 16;
  const int n0 = blockIdx.y * 64;

  v8f acc[4];
#pragma unroll
  for (int c = 0; c < 4; ++c)
#pragma unroll
    for (int i = 0; i < 8; ++i) acc[c][i] = 0.f;

  for (int k0 = 0; k0 < K; k0 += 32) {
    v16bf a = load_afrag(X + (size_t)m0 * K + k0, K, lane);
    v16bf bfr[4];
#pragma unroll
    for (int c = 0; c < 4; ++c)   // preload all B-fragments, then issue WMMAs
      bfr[c] = load_bfrag_t(W + (size_t)(n0 + c * 16) * K + k0, K, lane);
#pragma unroll
    for (int c = 0; c < 4; ++c)
      acc[c] = wmma_bf16(a, bfr[c], acc[c]);
  }

  const int half = lane >> 4, nl = lane & 15;
#pragma unroll
  for (int c = 0; c < 4; ++c) {
    const int col = n0 + c * 16 + nl;
    const float bv = bias[col];
#pragma unroll
    for (int r = 0; r < 8; ++r) {
      const int row = m0 + r + half * 8;
      const float v = acc[c][r] + bv;
      if (OUT_F32) ((float*)Yv)[(size_t)row * Nn + col] = v;
      else         ((__bf16*)Yv)[(size_t)row * Nn + col] = (__bf16)v;
    }
  }
}

// ---------------------------------------------------------------------------
// Flash attention with multiplicative sigmoid gate (exp2 domain):
//   softmax(QK^T*scale + log(g+eps)) V == normalize(exp2(v - m) * (g+eps)) V
//   with v = QK^T * scale * log2(e),  g = sigmoid(gw*SC + gb).
// Grid: (N/64, H, B). Block = 128 threads = 4 waves; wave w -> 16 query rows.
// K/V 32x64 tiles staged global->LDS by the Tensor Data Mover (TENSORcnt).
// ---------------------------------------------------------------------------
__global__ void __launch_bounds__(128)
flash_attn_wmma(const __bf16* __restrict__ Qp, const __bf16* __restrict__ Kp,
                const __bf16* __restrict__ Vp, const float* __restrict__ SC,
                const float* __restrict__ gw_p, const float* __restrict__ gb_p,
                __bf16* __restrict__ O) {
  __shared__ __bf16 sK[32 * DH];       // K tile  [key][d]   (TDM destination)
  __shared__ __bf16 sV[32 * DH];       // V tile  [key][d]   (TDM destination)
  __shared__ __bf16 sVt[DH * 32];      // V tile^T [d][key]
  __shared__ __bf16 sP[4 * 16 * 32];   // per-wave softmax probs (A-layout src)

  const int tid = threadIdx.x;
  const int lane = tid & 31, wave = tid >> 5;
  const int half = lane >> 4, nl = lane & 15;
  const int b = blockIdx.z, h = blockIdx.y;
  const int q0 = blockIdx.x * 64 + wave * 16;
  // exp2-domain gate scalars: sigmoid(x) = 1/(1+exp2(-(x*log2e)))
  const float gw2 = gw_p[0] * LOG2E, gb2 = gb_p[0] * LOG2E;
  const float k2 = SCALEF * LOG2E;

  const unsigned int ldsK = (unsigned int)(uintptr_t)(void*)sK;  // LDS byte offset
  const unsigned int ldsV = (unsigned int)(uintptr_t)(void*)sV;

  // Q fragments for this wave's 16 rows (d split into two 32-chunks)
  const __bf16* qbase = Qp + ((size_t)(b * NSEQ + q0)) * DM + h * DH;
  const v16bf qa0 = load_afrag(qbase, DM, lane);
  const v16bf qa1 = load_afrag(qbase + 32, DM, lane);

  v8f acc[4];
  float m_i[8], l_i[8];
#pragma unroll
  for (int c = 0; c < 4; ++c)
#pragma unroll
    for (int i = 0; i < 8; ++i) acc[c][i] = 0.f;
#pragma unroll
  for (int r = 0; r < 8; ++r) { m_i[r] = -INFINITY; l_i[r] = 0.f; }

  __bf16* myP = sP + wave * (16 * 32);
  const int srow = tid >> 2;          // 0..31 : transpose row (key)
  const int sd0 = (tid & 3) * 16;     // transpose d offset

  for (int k0 = 0; k0 < NSEQ; k0 += 32) {
    // ---- TDM: stage K and V 32x64 bf16 tiles into LDS (wave 0 issues) ----
    if (wave == 0) {
      const __bf16* kg = Kp + ((size_t)(b * NSEQ + k0)) * DM + h * DH;
      const __bf16* vg = Vp + ((size_t)(b * NSEQ + k0)) * DM + h * DH;
      tdm_load_tile_2d(ldsK, kg, DH, 32, DM);
      tdm_load_tile_2d(ldsV, vg, DH, 32, DM);
      __builtin_amdgcn_s_wait_tensorcnt(0);
    }
    if (k0 + 32 < NSEQ) {  // prefetch next tiles -> global_prefetch_b8
      const size_t nro = ((size_t)(b * NSEQ + k0 + 32 + srow)) * DM + h * DH + sd0;
      __builtin_prefetch(Kp + nro, 0, 1);
      __builtin_prefetch(Vp + nro, 0, 1);
    }
    __syncthreads();

    // ---- LDS->LDS transpose of V tile for the PV B-fragment layout ----
    {
      const __bf16* src = sV + srow * DH + sd0;
      v8bf vlo = *(const v8bf*)src;
      v8bf vhi = *(const v8bf*)(src + 8);
#pragma unroll
      for (int j = 0; j < 8; ++j) {
        sVt[(sd0 + j) * 32 + srow] = vlo[j];
        sVt[(sd0 + 8 + j) * 32 + srow] = vhi[j];
      }
    }
    __syncthreads();

    // ---- S = Q K^T (preload all 4 B-fragments, then 4 WMMAs) ----
    v16bf kb[4];
    kb[0] = load_bfrag_t(sK, DH, lane);
    kb[1] = load_bfrag_t(sK + 32, DH, lane);
    kb[2] = load_bfrag_t(sK + 16 * DH, DH, lane);
    kb[3] = load_bfrag_t(sK + 16 * DH + 32, DH, lane);
    v8f s0, s1;
#pragma unroll
    for (int i = 0; i < 8; ++i) { s0[i] = 0.f; s1[i] = 0.f; }
    s0 = wmma_bf16(qa0, kb[0], s0);
    s0 = wmma_bf16(qa1, kb[1], s0);
    s1 = wmma_bf16(qa0, kb[2], s1);
    s1 = wmma_bf16(qa1, kb[3], s1);

    // ---- gate (multiplicative) + online softmax in exp2 domain ----
    float v0[8], v1[8], g0a[8], g1a[8];
#pragma unroll
    for (int r = 0; r < 8; ++r) {
      const int q = q0 + r + half * 8;
      const float* sp = SC + ((size_t)b * NSEQ + q) * (size_t)NSEQ + k0;
      const float sc0 = sp[nl];
      const float sc1 = sp[16 + nl];
      g0a[r] = fast_rcp(1.f + fast_exp2(-(gw2 * sc0 + gb2))) + GEPS;
      g1a[r] = fast_rcp(1.f + fast_exp2(-(gw2 * sc1 + gb2))) + GEPS;
      v0[r] = s0[r] * k2;
      v1[r] = s1[r] * k2;
    }
#pragma unroll
    for (int r = 0; r < 8; ++r) {
      float rm = fmaxf(v0[r], v1[r]);
      rm = fmaxf(rm, __shfl_xor(rm, 1, 32));
      rm = fmaxf(rm, __shfl_xor(rm, 2, 32));
      rm = fmaxf(rm, __shfl_xor(rm, 4, 32));
      rm = fmaxf(rm, __shfl_xor(rm, 8, 32));
      const float newm = fmaxf(m_i[r], rm);
      const float alpha = fast_exp2(m_i[r] - newm);
      m_i[r] = newm;
      const float p0 = fast_exp2(v0[r] - newm) * g0a[r];
      const float p1 = fast_exp2(v1[r] - newm) * g1a[r];
      float rs = p0 + p1;
      rs += __shfl_xor(rs, 1, 32);
      rs += __shfl_xor(rs, 2, 32);
      rs += __shfl_xor(rs, 4, 32);
      rs += __shfl_xor(rs, 8, 32);
      l_i[r] = l_i[r] * alpha + rs;
#pragma unroll
      for (int c = 0; c < 4; ++c) acc[c][r] *= alpha;
      const int qloc = r + half * 8;
      myP[qloc * 32 + nl] = (__bf16)p0;
      myP[qloc * 32 + 16 + nl] = (__bf16)p1;
    }

    // ---- O += P V  (P via LDS for C->A relayout; preload B-frags) ----
    const v16bf pa = load_afrag(myP, 32, lane);
    v16bf vb[4];
#pragma unroll
    for (int c = 0; c < 4; ++c)
      vb[c] = load_bfrag_t(sVt + c * 16 * 32, 32, lane);
#pragma unroll
    for (int c = 0; c < 4; ++c)
      acc[c] = wmma_bf16(pa, vb[c], acc[c]);
    __syncthreads();
  }

  // ---- epilogue: normalize with v_rcp, store bf16 [B*N, 512] ----
#pragma unroll
  for (int r8 = 0; r8 < 8; ++r8) l_i[r8] = fast_rcp(l_i[r8]);
#pragma unroll
  for (int c = 0; c < 4; ++c) {
#pragma unroll
    for (int r = 0; r < 8; ++r) {
      const int q = q0 + r + half * 8;
      const int d = c * 16 + nl;
      const float ov = acc[c][r] * l_i[r];
      O[((size_t)(b * NSEQ + q)) * DM + h * DH + d] = (__bf16)ov;
    }
  }
}

// ---------------------------------------------------------------------------
// launch
// ---------------------------------------------------------------------------
extern "C" void kernel_launch(void* const* d_in, const int* in_sizes, int n_in,
                              void* d_out, int out_size, void* d_ws, size_t ws_size,
                              hipStream_t stream) {
  const float* Qin  = (const float*)d_in[0];
  const float* KVin = (const float*)d_in[1];
  const float* SC   = (const float*)d_in[2];
  const float* Wq   = (const float*)d_in[3];
  const float* Wqb  = (const float*)d_in[4];
  const float* Wk   = (const float*)d_in[5];
  const float* Wkb  = (const float*)d_in[6];
  const float* Wv   = (const float*)d_in[7];
  const float* Wvb  = (const float*)d_in[8];
  const float* gw   = (const float*)d_in[9];
  const float* gb   = (const float*)d_in[10];
  const float* Wo   = (const float*)d_in[11];
  const float* Wob  = (const float*)d_in[12];

  const int M = BB * NSEQ;  // 8192 rows

  // workspace carve (~52.4 MB)
  char* p = (char*)d_ws;
  auto take = [&](size_t bytes) -> void* {
    void* r = (void*)p;
    p += (bytes + 255) & ~(size_t)255;
    return r;
  };
  __bf16* Qin_bf = (__bf16*)take((size_t)M * DM * 2);
  __bf16* KV_bf  = (__bf16*)take((size_t)M * DM * 2);
  __bf16* Wq_bf  = (__bf16*)take((size_t)DM * DM * 2);
  __bf16* Wk_bf  = (__bf16*)take((size_t)DM * DM * 2);
  __bf16* Wv_bf  = (__bf16*)take((size_t)DM * DM * 2);
  __bf16* Wo_bf  = (__bf16*)take((size_t)DM * DM * 2);
  __bf16* Qp     = (__bf16*)take((size_t)M * DM * 2);
  __bf16* Kp     = (__bf16*)take((size_t)M * DM * 2);
  __bf16* Vp     = (__bf16*)take((size_t)M * DM * 2);
  __bf16* Obf    = (__bf16*)take((size_t)M * DM * 2);

  // 1) fp32 -> bf16 conversions
  const int nIn = M * DM;
  const int nW  = DM * DM;
  cvt_f32_bf16<<<2048, 256, 0, stream>>>(Qin,  Qin_bf, nIn);
  cvt_f32_bf16<<<2048, 256, 0, stream>>>(KVin, KV_bf,  nIn);
  cvt_f32_bf16<<<512,  256, 0, stream>>>(Wq, Wq_bf, nW);
  cvt_f32_bf16<<<512,  256, 0, stream>>>(Wk, Wk_bf, nW);
  cvt_f32_bf16<<<512,  256, 0, stream>>>(Wv, Wv_bf, nW);
  cvt_f32_bf16<<<512,  256, 0, stream>>>(Wo, Wo_bf, nW);

  // 2) Q/K/V projections (WMMA GEMM, bf16 out)
  dim3 ggrid(M / 128, DM / 64);
  gemm_xwT_wmma<false><<<ggrid, 256, 0, stream>>>(Qin_bf, Wq_bf, Wqb, Qp, M, DM, DM);
  gemm_xwT_wmma<false><<<ggrid, 256, 0, stream>>>(KV_bf,  Wk_bf, Wkb, Kp, M, DM, DM);
  gemm_xwT_wmma<false><<<ggrid, 256, 0, stream>>>(KV_bf,  Wv_bf, Wvb, Vp, M, DM, DM);

  // 3) flash attention with sigmoid gate (TDM-staged K/V tiles)
  dim3 agrid(NSEQ / 64, HEADS, BB);
  flash_attn_wmma<<<agrid, 128, 0, stream>>>(Qp, Kp, Vp, SC, gw, gb, Obf);

  // 4) output projection (WMMA GEMM, f32 out)
  gemm_xwT_wmma<true><<<ggrid, 256, 0, stream>>>(Obf, Wo_bf, Wob, (float*)d_out, M, DM, DM);
}
